// MambaMlpBlock_59863254172031
// MI455X (gfx1250) — matrix operand order repaired and checked by reading, not compiled, not run on hardware
//
#include <hip/hip_runtime.h>
#include <math.h>

#define D_MODEL 512
#define D_INNER 1024
#define D_STATE 16
#define DT_RANK 32
#define MLP_HID 2048
#define BATCH   4
#define SEQ     2048
#define NROWS   (BATCH * SEQ)   // 8192 token rows

typedef float          v8f    __attribute__((ext_vector_type(8)));
typedef unsigned int   u32x8  __attribute__((ext_vector_type(8)));
typedef __bf16         v16bf  __attribute__((ext_vector_type(16)));
typedef __bf16         bf16x4 __attribute__((ext_vector_type(4)));

enum { ACT_NONE = 0, ACT_SOFTPLUS = 1, ACT_GELU = 2 };

// ---------- helpers ----------
__device__ __forceinline__ v16bf frag_from_lds(const __bf16* p0, const __bf16* p1) {
  // two 16B LDS reads -> 16 bf16 fragment
  uint4 a = *(const uint4*)p0;
  uint4 b = *(const uint4*)p1;
  u32x8 t;
  t[0] = a.x; t[1] = a.y; t[2] = a.z; t[3] = a.w;
  t[4] = b.x; t[5] = b.y; t[6] = b.z; t[7] = b.w;
  return __builtin_bit_cast(v16bf, t);
}

// CDNA5 async global->LDS copy (ASYNCcnt-tracked), ISA 10.x / 15.18.3
__device__ __forceinline__ void async_copy_b128(unsigned lds_off, const void* gaddr) {
  asm volatile("global_load_async_to_lds_b128 %0, %1, off"
               :: "v"(lds_off), "v"(gaddr) : "memory");
}
__device__ __forceinline__ void wait_async0() {
  asm volatile("s_wait_asynccnt 0x0" ::: "memory");
}

// ---------- weight transpose + cast: src fp32 (N,K) -> dst bf16 (K, Nld) ----------
__global__ __launch_bounds__(256) void wcast_kernel(const float* __restrict__ src,
                                                    __bf16* __restrict__ dst,
                                                    int N, int K, int Nld) {
  int idx = blockIdx.x * 256 + threadIdx.x;
  int total = K * Nld;
  if (idx >= total) return;
  int k = idx / Nld;
  int n = idx - k * Nld;
  float v = (n < N) ? src[(size_t)n * K + k] : 0.f;
  dst[idx] = (__bf16)v;
}

// ---------- LayerNorm: one wave32 per 512-wide row ----------
__global__ __launch_bounds__(256) void layernorm_kernel(const float* __restrict__ x,
                                                        const float* __restrict__ w,
                                                        const float* __restrict__ b,
                                                        float* __restrict__ out) {
  int lane = threadIdx.x & 31;
  int row  = blockIdx.x * 8 + (threadIdx.x >> 5);
  const float* xr = x + (size_t)row * D_MODEL;
  float vals[16];
  float s = 0.f;
#pragma unroll
  for (int i = 0; i < 16; ++i) { vals[i] = xr[lane + 32 * i]; s += vals[i]; }
#pragma unroll
  for (int off = 16; off > 0; off >>= 1) s += __shfl_xor(s, off, 32);
  float mean = s * (1.f / (float)D_MODEL);
  float vs = 0.f;
#pragma unroll
  for (int i = 0; i < 16; ++i) { vals[i] -= mean; vs += vals[i] * vals[i]; }
#pragma unroll
  for (int off = 16; off > 0; off >>= 1) vs += __shfl_xor(vs, off, 32);
  float inv = rsqrtf(vs * (1.f / (float)D_MODEL) + 1e-5f);
  float* orow = out + (size_t)row * D_MODEL;
#pragma unroll
  for (int i = 0; i < 16; ++i) {
    int c = lane + 32 * i;
    orow[c] = vals[i] * inv * w[c] + b[c];
  }
}

// ---------- residual add + LayerNorm (x_mamba = x + dx; h2 = LN(x_mamba)) ----------
__global__ __launch_bounds__(256) void add_layernorm_kernel(const float* __restrict__ x,
                                                            const float* __restrict__ dx,
                                                            const float* __restrict__ w,
                                                            const float* __restrict__ b,
                                                            float* __restrict__ xm,
                                                            float* __restrict__ out) {
  int lane = threadIdx.x & 31;
  int row  = blockIdx.x * 8 + (threadIdx.x >> 5);
  size_t base = (size_t)row * D_MODEL;
  float vals[16];
  float s = 0.f;
#pragma unroll
  for (int i = 0; i < 16; ++i) {
    int c = lane + 32 * i;
    float v = x[base + c] + dx[base + c];
    xm[base + c] = v;
    vals[i] = v;
    s += v;
  }
#pragma unroll
  for (int off = 16; off > 0; off >>= 1) s += __shfl_xor(s, off, 32);
  float mean = s * (1.f / (float)D_MODEL);
  float vs = 0.f;
#pragma unroll
  for (int i = 0; i < 16; ++i) { vals[i] -= mean; vs += vals[i] * vals[i]; }
#pragma unroll
  for (int off = 16; off > 0; off >>= 1) vs += __shfl_xor(vs, off, 32);
  float inv = rsqrtf(vs * (1.f / (float)D_MODEL) + 1e-5f);
#pragma unroll
  for (int i = 0; i < 16; ++i) {
    int c = lane + 32 * i;
    out[base + c] = vals[i] * inv * w[c] + b[c];
  }
}

// ---------- depthwise causal conv (D_CONV=4) + SiLU over xs = xz[:, :, 0:1024] ----------
__global__ __launch_bounds__(256) void conv_silu_kernel(const float* __restrict__ xz,
                                                        const float* __restrict__ cw,
                                                        const float* __restrict__ cb,
                                                        float* __restrict__ u) {
  int idx = blockIdx.x * 256 + threadIdx.x;     // over NROWS * D_INNER
  int d  = idx & (D_INNER - 1);
  int bl = idx >> 10;                           // b*SEQ + l
  int l  = bl & (SEQ - 1);
  float acc = cb[d];
  const float* w4 = cw + d * 4;
#pragma unroll
  for (int t = 0; t < 4; ++t) {
    int ll = l - 3 + t;
    if (ll >= 0)
      acc += xz[(size_t)(bl - 3 + t) * (2 * D_INNER) + d] * w4[t];
  }
  float sig = 1.f / (1.f + __expf(-acc));
  u[idx] = acc * sig;
}

// ---------- selective scan (sequential over L), fused +u*D and *silu(z) ----------
// B/C staged 8 timesteps per barrier to amortize sync along the serial chain.
__global__ __launch_bounds__(256) void scan_kernel(const float* __restrict__ dt,
                                                   const float* __restrict__ u,
                                                   const float* __restrict__ xdbl,
                                                   const float* __restrict__ xz,
                                                   const float* __restrict__ A_log,
                                                   const float* __restrict__ Dp,
                                                   float* __restrict__ y) {
  int b = blockIdx.y;
  int d = blockIdx.x * 256 + threadIdx.x;
  float A[D_STATE];
#pragma unroll
  for (int n = 0; n < D_STATE; ++n) A[n] = -__expf(A_log[d * D_STATE + n]);
  float h[D_STATE];
#pragma unroll
  for (int n = 0; n < D_STATE; ++n) h[n] = 0.f;
  float Dpd = Dp[d];
  __shared__ float BC[8][2 * D_STATE];
  for (int l0 = 0; l0 < SEQ; l0 += 8) {
    {
      int step = threadIdx.x >> 5;
      int j    = threadIdx.x & 31;
      BC[step][j] = xdbl[((size_t)b * SEQ + l0 + step) * 128 + DT_RANK + j];
    }
    __syncthreads();
#pragma unroll
    for (int s = 0; s < 8; ++s) {
      size_t rb = (size_t)b * SEQ + l0 + s;
      float dtv = dt[rb * D_INNER + d];
      float uv  = u[rb * D_INNER + d];
      float zv  = xz[rb * (2 * D_INNER) + D_INNER + d];
      float acc = 0.f;
#pragma unroll
      for (int n = 0; n < D_STATE; ++n) {
        float dA = __expf(dtv * A[n]);
        h[n] = dA * h[n] + dtv * BC[s][n] * uv;
        acc += h[n] * BC[s][D_STATE + n];
      }
      float yv = acc + uv * Dpd;
      float sz = zv / (1.f + __expf(-zv));
      y[rb * D_INNER + d] = yv * sz;
    }
    __syncthreads();
  }
}

// ---------- WMMA bf16 GEMM: C(M,N) = A(M,K) fp32 * Wt(K,N) bf16, fused epilogue ----------
// Double-buffered LDS; A loaded to registers then cvt->bf16->LDS; B tile copied by the
// CDNA5 async global->LDS path (ASYNCcnt).
#define APITCH 40    // 32 + 8 halves pad
#define BPITCH 136   // 128 + 8 halves pad

__global__ __launch_bounds__(256) void gemm_wmma_bf16(const float* __restrict__ A, int lda,
                                                      const __bf16* __restrict__ Bw, int ldb,
                                                      const float* __restrict__ bias,
                                                      const float* __restrict__ resid,
                                                      float* __restrict__ C, int ldc,
                                                      int K, int act) {
  __shared__ __align__(16) __bf16 As[2][128 * APITCH];
  __shared__ __align__(16) __bf16 Bs[2][32 * BPITCH];
  int tid  = threadIdx.x;
  int lane = tid & 31;
  int wid  = tid >> 5;
  int wm   = wid & 3;   // 4 waves along M (32 rows each)
  int wn   = wid >> 2;  // 2 waves along N (64 cols each)
  int blockCol = blockIdx.x * 128;
  int blockRow = blockIdx.y * 128;

  v8f acc[2][4];
#pragma unroll
  for (int ms = 0; ms < 2; ++ms)
#pragma unroll
    for (int ns = 0; ns < 4; ++ns)
#pragma unroll
      for (int r = 0; r < 8; ++r) acc[ms][ns][r] = 0.f;

  float4 areg[4];
  auto loadA = [&](int k0) {
#pragma unroll
    for (int i = 0; i < 4; ++i) {
      int idx = tid + i * 256;          // 0..1023
      int row = idx >> 3;
      int c4  = (idx & 7) << 2;
      areg[i] = *(const float4*)(A + (size_t)(blockRow + row) * lda + k0 + c4);
    }
  };
  auto storeA = [&](int buf) {
#pragma unroll
    for (int i = 0; i < 4; ++i) {
      int idx = tid + i * 256;
      int row = idx >> 3;
      int c4  = (idx & 7) << 2;
      bf16x4 hh;
      hh[0] = (__bf16)areg[i].x; hh[1] = (__bf16)areg[i].y;
      hh[2] = (__bf16)areg[i].z; hh[3] = (__bf16)areg[i].w;
      *(bf16x4*)&As[buf][row * APITCH + c4] = hh;
    }
  };
  auto loadB_async = [&](int buf, int k0) {
#pragma unroll
    for (int i = 0; i < 2; ++i) {
      int idx = tid + i * 256;          // 0..511
      int row = idx >> 4;
      int col = (idx & 15) << 3;
      unsigned lds_off = (unsigned)(uintptr_t)&Bs[buf][row * BPITCH + col];
      async_copy_b128(lds_off, Bw + (size_t)(k0 + row) * ldb + blockCol + col);
    }
  };

  const int nk = K / 32;
  loadA(0);
  loadB_async(0, 0);
  storeA(0);
  wait_async0();
  __syncthreads();

  for (int t = 0; t < nk; ++t) {
    int cur = t & 1;
    int nxt = cur ^ 1;
    if (t + 1 < nk) {
      loadA((t + 1) * 32);
      loadB_async(nxt, (t + 1) * 32);
    }
    if (t + 2 < nk)
      __builtin_prefetch(A + (size_t)(blockRow + (tid >> 1)) * lda + (t + 2) * 32, 0, 1);

    v16bf aF[2];
#pragma unroll
    for (int ms = 0; ms < 2; ++ms) {
      int r   = wm * 32 + ms * 16 + (lane & 15);
      int off = (lane < 16) ? 0 : 8;    // ISA A-layout: lanes 0-15 K{0..7,16..23}, 16-31 K{8..15,24..31}
      aF[ms] = frag_from_lds(&As[cur][r * APITCH + off], &As[cur][r * APITCH + off + 16]);
    }
#pragma unroll
    for (int ns = 0; ns < 4; ++ns) {
      int n0 = wn * 64 + ns * 16;
      // B-layout: N = lane&15; lanes 0-15 K{0..7,16..23}, 16-31 K{8..15,24..31}
      v16bf bF;
#pragma unroll
      for (int s = 0; s < 16; ++s) {
        int kk = ((s < 8) ? s : (s + 8)) + ((lane < 16) ? 0 : 8);
        bF[s] = Bs[cur][kk * BPITCH + n0 + (lane & 15)];
      }
#pragma unroll
      for (int ms = 0; ms < 2; ++ms) {
        acc[ms][ns] = __builtin_amdgcn_wmma_f32_16x16x32_bf16(
            false, aF[ms], false, bF, (short)0, acc[ms][ns], false, false);
      }
    }
    if (t + 1 < nk) storeA(nxt);
    wait_async0();
    __syncthreads();
  }

  // epilogue: C/D layout — VGPR r: M = r (+8 for lanes 16-31), N = lane&15
#pragma unroll
  for (int ms = 0; ms < 2; ++ms) {
#pragma unroll
    for (int ns = 0; ns < 4; ++ns) {
      int col = blockCol + wn * 64 + ns * 16 + (lane & 15);
      float bv = bias ? bias[col] : 0.f;
#pragma unroll
      for (int r = 0; r < 8; ++r) {
        int row = blockRow + wm * 32 + ms * 16 + r + ((lane < 16) ? 0 : 8);
        float v = acc[ms][ns][r] + bv;
        if (act == ACT_SOFTPLUS)      v = (v > 20.f) ? v : log1pf(expf(v));
        else if (act == ACT_GELU)     v = 0.5f * v * (1.f + erff(v * 0.70710678118654752f));
        if (resid) v += resid[(size_t)row * ldc + col];
        C[(size_t)row * ldc + col] = v;
      }
    }
  }
}

// ---------- host launcher ----------
extern "C" void kernel_launch(void* const* d_in, const int* in_sizes, int n_in,
                              void* d_out, int out_size, void* d_ws, size_t ws_size,
                              hipStream_t stream) {
  (void)in_sizes; (void)n_in; (void)out_size; (void)ws_size;
  const float* x         = (const float*)d_in[0];
  const float* ln1_w     = (const float*)d_in[1];
  const float* ln1_b     = (const float*)d_in[2];
  const float* ln2_w     = (const float*)d_in[3];
  const float* ln2_b     = (const float*)d_in[4];
  const float* in_proj_w = (const float*)d_in[5];   // (2048, 512)
  const float* conv_w    = (const float*)d_in[6];   // (1024, 1, 4)
  const float* conv_b    = (const float*)d_in[7];
  const float* x_proj_w  = (const float*)d_in[8];   // (64, 1024)
  const float* dt_proj_w = (const float*)d_in[9];   // (1024, 32)
  const float* dt_proj_b = (const float*)d_in[10];
  const float* A_log     = (const float*)d_in[11];  // (1024, 16)
  const float* Dp        = (const float*)d_in[12];
  const float* out_projw = (const float*)d_in[13];  // (512, 1024)
  const float* fc1_w     = (const float*)d_in[14];  // (2048, 512)
  const float* fc1_b     = (const float*)d_in[15];
  const float* fc2_w     = (const float*)d_in[16];  // (512, 2048)
  const float* fc2_b     = (const float*)d_in[17];
  float* out = (float*)d_out;

  char* p = (char*)d_ws;
  auto carve = [&](size_t bytes) -> char* {
    char* r = p;
    p += (bytes + 255) & ~(size_t)255;
    return r;
  };
  __bf16* WtIn = (__bf16*)carve((size_t)512 * 2048 * 2);
  __bf16* WtXp = (__bf16*)carve((size_t)1024 * 128 * 2);
  __bf16* WtDt = (__bf16*)carve((size_t)32 * 1024 * 2);
  __bf16* WtOp = (__bf16*)carve((size_t)1024 * 512 * 2);
  __bf16* WtF1 = (__bf16*)carve((size_t)512 * 2048 * 2);
  __bf16* WtF2 = (__bf16*)carve((size_t)2048 * 512 * 2);
  float* h1    = (float*)carve((size_t)NROWS * D_MODEL * 4);
  float* xz    = (float*)carve((size_t)NROWS * 2 * D_INNER * 4);   // also reused as fc1 output
  float* u     = (float*)carve((size_t)NROWS * D_INNER * 4);       // also reused as out_proj output
  float* xdbl  = (float*)carve((size_t)NROWS * 128 * 4);
  float* dtbuf = (float*)carve((size_t)NROWS * D_INNER * 4);       // also reused as h2
  float* ybuf  = (float*)carve((size_t)NROWS * D_INNER * 4);
  float* xm    = (float*)carve((size_t)NROWS * D_MODEL * 4);
  float* fc1o  = xz;      // alias: xz dead after scan
  float* oproj = u;       // alias: u dead after scan
  float* h2    = dtbuf;   // alias: dt dead after scan

  auto wcast = [&](const float* src, __bf16* dst, int N, int K, int Nld) {
    int total = K * Nld;
    wcast_kernel<<<(total + 255) / 256, 256, 0, stream>>>(src, dst, N, K, Nld);
  };
  // weight prep (fp32 (N,K) -> bf16 (K,Nld))
  wcast(in_proj_w, WtIn, 2048, 512, 2048);
  wcast(x_proj_w,  WtXp, 64,   1024, 128);
  wcast(dt_proj_w, WtDt, 1024, 32,   1024);
  wcast(out_projw, WtOp, 512,  1024, 512);
  wcast(fc1_w,     WtF1, 2048, 512,  2048);
  wcast(fc2_w,     WtF2, 512,  2048, 512);

  const int rowBlocks = NROWS / 8;   // wave-per-row LN kernels
  // 1) h1 = LN(x)
  layernorm_kernel<<<rowBlocks, 256, 0, stream>>>(x, ln1_w, ln1_b, h1);
  // 2) xz = h1 @ in_proj_w^T    (8192 x 2048, K=512)
  gemm_wmma_bf16<<<dim3(2048 / 128, NROWS / 128), 256, 0, stream>>>(
      h1, D_MODEL, WtIn, 2048, nullptr, nullptr, xz, 2048, 512, ACT_NONE);
  // 3) u = silu(causal depthwise conv(xs) + conv_b)
  conv_silu_kernel<<<(NROWS * D_INNER) / 256, 256, 0, stream>>>(xz, conv_w, conv_b, u);
  // 4) x_dbl = u @ x_proj_w^T   (8192 x 128 padded, K=1024)
  gemm_wmma_bf16<<<dim3(1, NROWS / 128), 256, 0, stream>>>(
      u, D_INNER, WtXp, 128, nullptr, nullptr, xdbl, 128, 1024, ACT_NONE);
  // 5) dt = softplus(dt_r @ dt_proj_w^T + dt_proj_b)   (8192 x 1024, K=32; A = x_dbl cols 0..31)
  gemm_wmma_bf16<<<dim3(1024 / 128, NROWS / 128), 256, 0, stream>>>(
      xdbl, 128, WtDt, 1024, dt_proj_b, nullptr, dtbuf, 1024, 32, ACT_SOFTPLUS);
  // 6) selective scan (+ u*D, * silu(z)) -> y
  scan_kernel<<<dim3(D_INNER / 256, BATCH), 256, 0, stream>>>(
      dtbuf, u, xdbl, xz, A_log, Dp, ybuf);
  // 7) oproj = y @ out_proj_w^T (8192 x 512, K=1024)
  gemm_wmma_bf16<<<dim3(512 / 128, NROWS / 128), 256, 0, stream>>>(
      ybuf, D_INNER, WtOp, 512, nullptr, nullptr, oproj, 512, 1024, ACT_NONE);
  // 8) xm = x + oproj; h2 = LN(xm)
  add_layernorm_kernel<<<rowBlocks, 256, 0, stream>>>(x, oproj, ln2_w, ln2_b, xm, h2);
  // 9) fc1o = gelu(h2 @ fc1_w^T + fc1_b)   (8192 x 2048, K=512)
  gemm_wmma_bf16<<<dim3(2048 / 128, NROWS / 128), 256, 0, stream>>>(
      h2, D_MODEL, WtF1, 2048, fc1_b, nullptr, fc1o, 2048, 512, ACT_GELU);
  // 10) out = xm + fc1o @ fc2_w^T + fc2_b  (8192 x 512, K=2048)
  gemm_wmma_bf16<<<dim3(512 / 128, NROWS / 128), 256, 0, stream>>>(
      fc1o, 2048, WtF2, 512, fc2_b, xm, out, 512, 2048, ACT_NONE);
}
